// GATClassifier_36618891165995
// MI455X (gfx1250) — matrix operand order repaired and checked by reading, not compiled, not run on hardware
//
#include <hip/hip_runtime.h>

// ---------------------------------------------------------------------------
// Types for CDNA5 WMMA (wave32, 16x16x32 bf16 -> f32)
// ---------------------------------------------------------------------------
typedef __attribute__((ext_vector_type(16))) __bf16 v16bf;
typedef __attribute__((ext_vector_type(2)))  __bf16 v2bf;
typedef __attribute__((ext_vector_type(8)))  float  v8f;

// Order-preserving float<->uint mapping so we can use integer atomicMax for
// the segment-max (edge softmax) pass.
__device__ __forceinline__ unsigned floatFlip(float f) {
    unsigned u = __float_as_uint(f);
    return (u & 0x80000000u) ? ~u : (u | 0x80000000u);
}
__device__ __forceinline__ float floatUnflip(unsigned u) {
    unsigned v = (u & 0x80000000u) ? (u & 0x7fffffffu) : ~u;
    return __uint_as_float(v);
}
// floatFlip(-inf) == 0x007FFFFF : init value for the max buffers.
#define NEG_INF_FLIPPED 0x007FFFFFu

// ---------------------------------------------------------------------------
// Tiled WMMA GEMM: C[M,Ncols] = A[M,K] @ B[K,Ncols]
//   - 256 threads = 8 waves per block; each wave owns one 16-row M tile and
//     computes a 16x64 output strip (NT=4 N-tiles, 4 accumulators).
//   - B panel staged K-major (transposed) in LDS as bf16 with +16 element
//     column padding: every wave fragment is 16 contiguous bf16 at a
//     32B-aligned address -> 2x ds_load_b128 per fragment (no scalar merges).
//   - A fragment per-lane K-chunks are contiguous 16-float runs ->
//     4x global_load_b128 per k-step.
//   - K template parameter (128 / 256) fully unrolls the k-loop:
//     K/32 * 4 static v_wmma per instantiation.
// Fragment layouts (A/B share contiguous k = 16*half + s; 16-bit A 16x32; 16-bit B 32x16;
// 32-bit C/D 16x16). EXEC stays all-ones around every WMMA.
// ---------------------------------------------------------------------------
template <int K, int NT>
__global__ __launch_bounds__(256)
void wmma_gemm_tiled(const float* __restrict__ A,
                     const float* __restrict__ B,
                     float* __restrict__ C,
                     int Mtiles, int Ncols) {
    constexpr int KP = K + 16;              // padded column stride (elements);
    __shared__ __bf16 Bs[64 * KP];          // KP*2 bytes = multiple of 32B

    const int tid     = threadIdx.x;
    const int colBase = blockIdx.y * 64;

    // ---- stage B panel [K x 64] -> LDS transposed: Bs[c*KP + k] (bf16) ----
    // Threads sweep columns of two adjacent B rows (coalesced global reads),
    // pack a (k, k+1) pair and store it with one b32 LDS store.
    for (int idx = tid; idx < (K / 2) * 64; idx += 256) {
        int kk = idx >> 6;                  // pair index
        int c  = idx & 63;
        int k  = kk * 2;
        v2bf p;
        p[0] = (__bf16)B[(size_t)k       * Ncols + colBase + c];
        p[1] = (__bf16)B[(size_t)(k + 1) * Ncols + colBase + c];
        *(v2bf*)&Bs[c * KP + k] = p;
    }
    __syncthreads();

    const int wid  = tid >> 5;
    const int lane = tid & 31;
    const int half = lane >> 4;   // 0 or 1
    const int l16  = lane & 15;

    int mtile = blockIdx.x * 8 + wid;
    if (mtile >= Mtiles) mtile = Mtiles - 1;  // duplicate tile: identical writes
    const float* __restrict__ arow = A + ((size_t)mtile * 16 + l16) * K;

    v8f acc[NT] = {};

#pragma unroll
    for (int k0 = 0; k0 < K; k0 += 32) {
        // ---- A fragment: lanes 0-15 hold K=k0+{0..15},
        //                  lanes 16-31 hold K=k0+{16..31}
        // Each lane reads 16 contiguous floats -> 4x float4 (b128) loads.
        const float4 f0 = *(const float4*)(arow + k0 + half * 16);
        const float4 f1 = *(const float4*)(arow + k0 + half * 16 + 4);
        const float4 f2 = *(const float4*)(arow + k0 + half * 16 + 8);
        const float4 f3 = *(const float4*)(arow + k0 + half * 16 + 12);
        v16bf a;
        a[0]  = (__bf16)f0.x;  a[1]  = (__bf16)f0.y;
        a[2]  = (__bf16)f0.z;  a[3]  = (__bf16)f0.w;
        a[4]  = (__bf16)f1.x;  a[5]  = (__bf16)f1.y;
        a[6]  = (__bf16)f1.z;  a[7]  = (__bf16)f1.w;
        a[8]  = (__bf16)f2.x;  a[9]  = (__bf16)f2.y;
        a[10] = (__bf16)f2.z;  a[11] = (__bf16)f2.w;
        a[12] = (__bf16)f3.x;  a[13] = (__bf16)f3.y;
        a[14] = (__bf16)f3.z;  a[15] = (__bf16)f3.w;

        // ---- 4 N-tiles share this A fragment. B fragment for (nt, lane) is
        //      16 contiguous bf16 at Bs[col*KP + k0 + half*16] (32B aligned).
#pragma unroll
        for (int nt = 0; nt < NT; ++nt) {
            const int col = nt * 16 + l16;
            const v16bf b = *(const v16bf*)&Bs[(size_t)col * KP + k0 + half * 16];
            acc[nt] = __builtin_amdgcn_wmma_f32_16x16x32_bf16(
                /*neg_a=*/false, a, /*neg_b=*/false, b,
                /*c_mod=*/(short)0, acc[nt], /*reuse_a=*/false, /*reuse_b=*/false);
        }
    }

    // ---- C/D layout: VGPR i -> row = mtile*16 + half*8 + i
#pragma unroll
    for (int nt = 0; nt < NT; ++nt) {
#pragma unroll
        for (int i = 0; i < 8; ++i) {
            int r = mtile * 16 + half * 8 + i;
            C[(size_t)r * Ncols + colBase + nt * 16 + l16] = acc[nt][i];
        }
    }
}

// ---------------------------------------------------------------------------
// alpha_s[n,h] = sum_c h[n,h,c]*a_src[h,c]; alpha_d likewise. C = 64 fixed.
// ---------------------------------------------------------------------------
__global__ void alpha_kernel(const float* __restrict__ h,
                             const float* __restrict__ a_src,
                             const float* __restrict__ a_dst,
                             float* __restrict__ as, float* __restrict__ ad,
                             int Nn, int H) {
    int t = blockIdx.x * blockDim.x + threadIdx.x;
    if (t >= Nn * H) return;
    int hh = t % H;
    const float* hp  = h + (size_t)t * 64;
    const float* asp = a_src + hh * 64;
    const float* adp = a_dst + hh * 64;
    float s = 0.f, d = 0.f;
#pragma unroll 4
    for (int c = 0; c < 64; ++c) {
        float v = hp[c];
        s += v * asp[c];
        d += v * adp[c];
    }
    as[t] = s;
    ad[t] = d;
}

// ---------------------------------------------------------------------------
// Edge pass 1: segment max of leaky_relu(alpha_s[src]+alpha_d[dst]) per dst.
// Edge list = concat(edge_index, self loops). e in [E, E+Nn) -> s=d=e-E.
// ---------------------------------------------------------------------------
__global__ void edge_attn_max(const int* __restrict__ ei, int E, int Nn, int H,
                              const float* __restrict__ as,
                              const float* __restrict__ ad,
                              unsigned* __restrict__ m) {
    int e = blockIdx.x * blockDim.x + threadIdx.x;
    if (e >= E + Nn) return;
    int s, d;
    if (e < E) { s = ei[e]; d = ei[E + e]; } else { s = d = e - E; }
    for (int h = 0; h < H; ++h) {
        float v = as[s * H + h] + ad[d * H + h];
        v = v > 0.f ? v : 0.2f * v;
        atomicMax(&m[d * H + h], floatFlip(v));
    }
}

// Edge pass 2: denom[dst,h] += exp(e - m[dst,h])
__global__ void edge_attn_sum(const int* __restrict__ ei, int E, int Nn, int H,
                              const float* __restrict__ as,
                              const float* __restrict__ ad,
                              const unsigned* __restrict__ m,
                              float* __restrict__ den) {
    int e = blockIdx.x * blockDim.x + threadIdx.x;
    if (e >= E + Nn) return;
    int s, d;
    if (e < E) { s = ei[e]; d = ei[E + e]; } else { s = d = e - E; }
    for (int h = 0; h < H; ++h) {
        float v = as[s * H + h] + ad[d * H + h];
        v = v > 0.f ? v : 0.2f * v;
        float mx = floatUnflip(m[d * H + h]);
        atomicAdd(&den[d * H + h], __expf(v - mx));
    }
}

// Edge pass 3: out[dst,h,:] += w * h[src,h,:]  (one wave per (edge, head))
__global__ void edge_aggregate(const int* __restrict__ ei, int E, int Nn, int H,
                               const float* __restrict__ as,
                               const float* __restrict__ ad,
                               const unsigned* __restrict__ m,
                               const float* __restrict__ den,
                               const float* __restrict__ hfeat,
                               float* __restrict__ out) {
    int gtid = blockIdx.x * blockDim.x + threadIdx.x;
    int wave = gtid >> 5;
    int lane = gtid & 31;
    long total = (long)(E + Nn) * H;
    if ((long)wave >= total) return;
    int e = wave / H;
    int h = wave - e * H;
    int s, d;
    if (e < E) { s = ei[e]; d = ei[E + e]; } else { s = d = e - E; }
    float v = as[s * H + h] + ad[d * H + h];
    v = v > 0.f ? v : 0.2f * v;
    float mx = floatUnflip(m[d * H + h]);
    float w = __expf(v - mx) / (den[d * H + h] + 1e-16f);
    const float* hp = hfeat + ((size_t)s * H + h) * 64;
    float* op       = out   + ((size_t)d * H + h) * 64;
    atomicAdd(&op[lane],       w * hp[lane]);
    atomicAdd(&op[lane + 32],  w * hp[lane + 32]);
}

// out[t] = elu(in[t] + bias[t % C])
__global__ void bias_elu(const float* __restrict__ in,
                         const float* __restrict__ bias,
                         float* __restrict__ out, int total, int Cc) {
    int t = blockIdx.x * blockDim.x + threadIdx.x;
    if (t >= total) return;
    float v = in[t] + bias[t % Cc];
    out[t] = v > 0.f ? v : (__expf(v) - 1.f);
}

// global mean-pool accumulation (one wave per node, 64 channels)
__global__ void pool_kernel(const float* __restrict__ h,
                            const int* __restrict__ batch,
                            float* __restrict__ pool, float* __restrict__ cnt,
                            int Nn) {
    int gtid = blockIdx.x * blockDim.x + threadIdx.x;
    int node = gtid >> 5;
    int lane = gtid & 31;
    if (node >= Nn) return;
    int b = batch[node];
    const float* hp = h + (size_t)node * 64;
    atomicAdd(&pool[b * 64 + lane],      hp[lane]);
    atomicAdd(&pool[b * 64 + lane + 32], hp[lane + 32]);
    if (lane == 0) atomicAdd(&cnt[b], 1.0f);
}

// logits[b] = (pool[b,:]/max(cnt,1)) . lin_w + lin_b
__global__ void final_linear(const float* __restrict__ pool,
                             const float* __restrict__ cnt,
                             const float* __restrict__ lw,
                             const float* __restrict__ lb,
                             float* __restrict__ out, int Bb) {
    int b = blockIdx.x * blockDim.x + threadIdx.x;
    if (b >= Bb) return;
    float c = cnt[b];
    c = c > 1.f ? c : 1.f;
    float s = 0.f;
#pragma unroll 4
    for (int i = 0; i < 64; ++i) s += (pool[b * 64 + i] / c) * lw[i];
    out[b] = s + lb[0];
}

__global__ void fill_f32(float* p, float v, int n) {
    int t = blockIdx.x * blockDim.x + threadIdx.x;
    if (t < n) p[t] = v;
}
__global__ void fill_u32(unsigned* p, unsigned v, int n) {
    int t = blockIdx.x * blockDim.x + threadIdx.x;
    if (t < n) p[t] = v;
}

// ---------------------------------------------------------------------------
// Host-side launcher
// ---------------------------------------------------------------------------
static inline int cdiv(long a, long b) { return (int)((a + b - 1) / b); }

extern "C" void kernel_launch(void* const* d_in, const int* in_sizes, int n_in,
                              void* d_out, int out_size, void* d_ws, size_t ws_size,
                              hipStream_t stream) {
    const float* x      = (const float*)d_in[0];   // [N,128]
    const int*   ei     = (const int*)  d_in[1];   // [2,E]
    const int*   batch  = (const int*)  d_in[2];   // [N]
    const float* W1     = (const float*)d_in[3];   // [128,256]
    const float* a1s    = (const float*)d_in[4];   // [4,64]
    const float* a1d    = (const float*)d_in[5];   // [4,64]
    const float* b1     = (const float*)d_in[6];   // [256]
    const float* W2     = (const float*)d_in[7];   // [256,64]
    const float* a2s    = (const float*)d_in[8];   // [1,64]
    const float* a2d    = (const float*)d_in[9];   // [1,64]
    const float* b2     = (const float*)d_in[10];  // [64]
    const float* lw     = (const float*)d_in[11];  // [64,1]
    const float* lb     = (const float*)d_in[12];  // [1]
    float* out = (float*)d_out;                    // [B]

    const int Nn = in_sizes[0] / 128;   // 50000 (divisible by 16)
    const int E  = in_sizes[1] / 2;     // 1000000
    const int Et = E + Nn;              // edges incl. self loops
    const int Bb = out_size;            // 64
    const int Mtiles = Nn / 16;         // 3125

    // ---- workspace partition (floats) -------------------------------------
    float* ws = (float*)d_ws;
    float*    h1   = ws;                                 // N*256
    float*    agg1 = h1   + (size_t)Nn * 256;            // N*256 (-> x2 after elu)
    float*    as1  = agg1 + (size_t)Nn * 256;            // N*4
    float*    ad1  = as1  + (size_t)Nn * 4;              // N*4
    unsigned* m1   = (unsigned*)(ad1 + (size_t)Nn * 4);  // N*4
    float*    den1 = (float*)m1 + (size_t)Nn * 4;        // N*4
    float*    as2  = den1 + (size_t)Nn * 4;              // N
    float*    ad2  = as2  + Nn;                          // N
    unsigned* m2   = (unsigned*)(ad2 + Nn);              // N
    float*    den2 = (float*)m2 + Nn;                    // N
    float*    pool = den2 + Nn;                          // B*64
    float*    cnt  = pool + (size_t)Bb * 64;             // B
    // h1 region is dead after layer-1 aggregation: reuse it for layer 2.
    float*    h2   = h1;                                 // N*64
    float*    agg2 = h1 + (size_t)Nn * 64;               // N*64

    const int TB = 256;
    const int gemmBlocksX = cdiv(Mtiles, 8);  // 8 waves (M tiles) per block

    // ==================== Layer 1 (4 heads, 64 ch) =========================
    // h1 = x @ W1   [N,128]@[128,256]
    wmma_gemm_tiled<128, 4><<<dim3(gemmBlocksX, 256 / 64), 256, 0, stream>>>(
        x, W1, h1, Mtiles, 256);
    // alpha
    alpha_kernel<<<cdiv((long)Nn * 4, TB), TB, 0, stream>>>(h1, a1s, a1d, as1, ad1, Nn, 4);
    // init max/denom/agg
    fill_u32<<<cdiv((long)Nn * 4, TB), TB, 0, stream>>>(m1, NEG_INF_FLIPPED, Nn * 4);
    fill_f32<<<cdiv((long)Nn * 4, TB), TB, 0, stream>>>(den1, 0.f, Nn * 4);
    fill_f32<<<cdiv((long)Nn * 256, TB), TB, 0, stream>>>(agg1, 0.f, Nn * 256);
    // edge softmax + aggregate
    edge_attn_max<<<cdiv(Et, TB), TB, 0, stream>>>(ei, E, Nn, 4, as1, ad1, m1);
    edge_attn_sum<<<cdiv(Et, TB), TB, 0, stream>>>(ei, E, Nn, 4, as1, ad1, m1, den1);
    edge_aggregate<<<cdiv((long)Et * 4, 8), TB, 0, stream>>>(ei, E, Nn, 4, as1, ad1,
                                                             m1, den1, h1, agg1);
    // x2 = elu(agg1 + b1)
    bias_elu<<<cdiv((long)Nn * 256, TB), TB, 0, stream>>>(agg1, b1, agg1, Nn * 256, 256);

    // ==================== Layer 2 (1 head, 64 ch) ==========================
    // h2 = x2 @ W2   [N,256]@[256,64]   (h2 overlays dead h1 region)
    wmma_gemm_tiled<256, 4><<<dim3(gemmBlocksX, 1), 256, 0, stream>>>(
        agg1, W2, h2, Mtiles, 64);
    alpha_kernel<<<cdiv(Nn, TB), TB, 0, stream>>>(h2, a2s, a2d, as2, ad2, Nn, 1);
    fill_u32<<<cdiv(Nn, TB), TB, 0, stream>>>(m2, NEG_INF_FLIPPED, Nn);
    fill_f32<<<cdiv(Nn, TB), TB, 0, stream>>>(den2, 0.f, Nn);
    fill_f32<<<cdiv((long)Nn * 64, TB), TB, 0, stream>>>(agg2, 0.f, Nn * 64);
    edge_attn_max<<<cdiv(Et, TB), TB, 0, stream>>>(ei, E, Nn, 1, as2, ad2, m2);
    edge_attn_sum<<<cdiv(Et, TB), TB, 0, stream>>>(ei, E, Nn, 1, as2, ad2, m2, den2);
    edge_aggregate<<<cdiv((long)Et, 8), TB, 0, stream>>>(ei, E, Nn, 1, as2, ad2,
                                                         m2, den2, h2, agg2);
    bias_elu<<<cdiv((long)Nn * 64, TB), TB, 0, stream>>>(agg2, b2, agg2, Nn * 64, 64);

    // ==================== Pool + classifier ================================
    fill_f32<<<cdiv((long)Bb * 64 + Bb, TB), TB, 0, stream>>>(pool, 0.f, Bb * 64 + Bb);
    pool_kernel<<<cdiv((long)Nn * 32, TB), TB, 0, stream>>>(agg2, batch, pool, cnt, Nn);
    final_linear<<<1, Bb, 0, stream>>>(pool, cnt, lw, lb, out, Bb);
}
